// DilateResNetBidirection_7859790152069
// MI455X (gfx1250) — compile-verified, matrix operand order checked by reading, not run on hardware
//
#include <hip/hip_runtime.h>
#include <hip/hip_bf16.h>

// ---------------------------------------------------------------------------
// DilateResNetBidirection on gfx1250 (MI455X): implicit-GEMM 3x3 convs via
// v_wmma_f32_16x16x32_bf16. NHWC bf16 activations. Per-layer weight fragments
// staged into LDS with global_load_async_to_lds_b128 (ASYNCcnt) and consumed
// via ds_load_b128. A-fragment loads software-pipelined across taps.
// ---------------------------------------------------------------------------

typedef __attribute__((ext_vector_type(16))) __bf16 v16bf;
typedef __attribute__((ext_vector_type(8)))  float  v8f;

#define HH   256
#define WW   256
#define HWC  (HH * WW)
#define BB   4
#define BHW  (BB * HWC)
#define NHID 64
#define NCLS 25
#define NBLK 10

union AFrag { uint4 q[2]; v16bf v; };

// ---------------------------------------------------------------------------
// Implicit-GEMM 3x3 conv. 8 waves/WG; one wave -> 16 pixels x (NG*16) ch.
// MODE 0: plain bf16 NHWC out
// MODE 1: out = relu(scale*out + shift) -> bf16 NHWC   (fused BN2+ReLU)
// MODE 2: out += resid -> bf16 NHWC                    (fused residual add)
// MODE 3: out += bias  -> f32 NCHW, only oc < OCvalid  (final conv)
// DUAL: second 9-tap pass (dilation dil2) with weights WF2.
// ---------------------------------------------------------------------------
template <int NG, int KC, int MODE, bool DUAL>
__global__ __launch_bounds__(256)
void conv3x3_wmma_kernel(const __bf16* __restrict__ X,
                         const __bf16* __restrict__ WF1,
                         const __bf16* __restrict__ WF2, int dil2,
                         const float*  __restrict__ scale,
                         const float*  __restrict__ shift,
                         const __bf16* __restrict__ resid,
                         const float*  __restrict__ bias, int OCvalid,
                         __bf16* __restrict__ Ybf,
                         float*  __restrict__ Yf32,
                         const __bf16* __restrict__ zpage)
{
    constexpr int CSin  = KC * 32;
    constexpr int OCpad = NG * 16;
    constexpr int FRAG  = NG * 9 * KC * 512;         // bf16 elems per weight set
    constexpr int TOT   = (DUAL ? 2 : 1) * FRAG;     // bf16 elems staged in LDS
    constexpr int NCHNK = (TOT * 2) / 16;            // b128 chunks

    __shared__ __bf16 smw[TOT];

    // ---- stage weight fragments into LDS via async global->LDS DMA --------
    {
        const unsigned int lbase = (unsigned int)(uintptr_t)(&smw[0]);
        for (int c = threadIdx.x; c < NCHNK; c += 256) {
            const __bf16* gp = (c < FRAG / 8) ? (WF1 + (c << 3))
                                              : (WF2 + ((c - FRAG / 8) << 3));
            unsigned long long ga = (unsigned long long)(uintptr_t)gp;
            unsigned int loff = lbase + ((unsigned int)c << 4);
            asm volatile("global_load_async_to_lds_b128 %0, %1, off"
                         :: "v"(loff), "v"(ga) : "memory");
        }
        asm volatile("s_wait_asynccnt 0x0" ::: "memory");
    }
    __syncthreads();

    const int lane = threadIdx.x & 31;
    const int wave = threadIdx.x >> 5;
    const int p0   = (blockIdx.x * 8 + wave) << 4;   // 16 pixels per wave
    const int row  = lane & 15;
    const int hi   = lane >> 4;                      // K-half select (A layout)
    const int p    = p0 + row;
    const int x    = p & (WW - 1);
    const int y    = (p >> 8) & (HH - 1);
    const int b    = p >> 16;

    // A source address for (pass, tap); zero page when out of bounds
    auto a_src = [&](int pass, int tap) -> const __bf16* {
        const int dil = pass ? dil2 : 1;
        const int yy  = y + (tap / 3 - 1) * dil;
        const int xx  = x + (tap % 3 - 1) * dil;
        const bool oob = ((unsigned)yy >= (unsigned)HH) |
                         ((unsigned)xx >= (unsigned)WW);
        return oob ? zpage : X + (size_t)((b * HH + yy) * WW + xx) * CSin;
    };

    v8f acc[NG] = {};

    constexpr int PASSES = DUAL ? 2 : 1;
    constexpr int STEPS  = PASSES * 9 * KC;

    // software pipeline: load A for step s+1 while issuing WMMAs of step s
    AFrag a_cur;
    {
        const __bf16* sp = a_src(0, 0);
        const int kb = hi * 8;
        a_cur.q[0] = *(const uint4*)(sp + kb);
        a_cur.q[1] = *(const uint4*)(sp + kb + 16);
    }
#pragma unroll
    for (int s = 0; s < STEPS; ++s) {
        const int pass = s / (9 * KC);
        const int rem  = s % (9 * KC);
        const int tap  = rem / KC;
        const int kc   = rem % KC;

        AFrag a_nxt = a_cur;
        if (s + 1 < STEPS) {
            const int s2    = s + 1;
            const int pass2 = s2 / (9 * KC);
            const int rem2  = s2 % (9 * KC);
            const int tap2  = rem2 / KC;
            const int kc2   = rem2 % KC;
            const __bf16* sp = a_src(pass2, tap2);
            const int kb = kc2 * 32 + hi * 8;
            a_nxt.q[0] = *(const uint4*)(sp + kb);
            a_nxt.q[1] = *(const uint4*)(sp + kb + 16);
        }

        const __bf16* wb = smw + pass * FRAG;        // LDS (ds_load_b128)
#pragma unroll
        for (int g = 0; g < NG; ++g) {
            const uint4* wp = (const uint4*)(wb + ((((g * 9 + tap) * KC) + kc) << 9)
                                                + (lane << 4));
            AFrag w;
            w.q[0] = wp[0];
            w.q[1] = wp[1];
            acc[g] = __builtin_amdgcn_wmma_f32_16x16x32_bf16(
                false, a_cur.v, false, w.v, (short)0, acc[g], false, false);
        }
        a_cur = a_nxt;
    }

    // Epilogue. C/D layout: lanes 0-15: N=lane, M=r; lanes 16-31: N=lane-16, M=8+r.
    const int n     = lane & 15;
    const int mbase = hi << 3;
#pragma unroll
    for (int g = 0; g < NG; ++g) {
        const int oc = g * 16 + n;
#pragma unroll
        for (int r = 0; r < 8; ++r) {
            const int pp = p0 + mbase + r;
            float v = acc[g][r];
            if (MODE == 1) v = fmaxf(v * scale[oc] + shift[oc], 0.0f);
            if (MODE == 2) v += (float)resid[(size_t)pp * OCpad + oc];
            if (MODE == 3) {
                if (oc < OCvalid) {
                    v += bias[oc];
                    const int xb = pp & (WW - 1);
                    const int yb = (pp >> 8) & (HH - 1);
                    const int bb = pp >> 16;
                    Yf32[(size_t)(bb * OCvalid + oc) * HWC + (yb << 8) + xb] = v;
                }
            } else {
                Ybf[(size_t)pp * OCpad + oc] = (__bf16)v;
            }
        }
    }
}

// ---------------------------------------------------------------------------
// Weight repack: (O,I,3,3) f32 -> bf16 B-fragments [g][tap][kc][lane][16],
// matching the per-lane 16-bit B layout (mirror of A). Zero-pads O and I.
// ---------------------------------------------------------------------------
__global__ void repack_weights_kernel(const float* __restrict__ Wsrc,
                                      __bf16* __restrict__ frag,
                                      int O, int I, int ngroups, int kchunks)
{
    const int idx   = blockIdx.x * 256 + threadIdx.x;
    const int total = ngroups * 9 * kchunks * 512;
    if (idx >= total) return;
    const int e    = idx & 15;
    const int l    = (idx >> 4) & 31;
    int rest       = idx >> 9;
    const int kc   = rest % kchunks; rest /= kchunks;
    const int tap  = rest % 9;
    const int g    = rest / 9;
    const int N    = l & 15;
    const int hi   = (l >> 4) & 1;
    const int K    = (e < 8) ? (hi * 8 + e) : (16 + hi * 8 + (e - 8));
    const int o    = g * 16 + N;
    const int i    = kc * 32 + K;
    float v = 0.0f;
    if (o < O && i < I) v = Wsrc[(size_t)(o * I + i) * 9 + tap];
    frag[idx] = (__bf16)v;
}

// BN inference params: scale = g*rsqrt(v+eps); shift = b - m*scale
__global__ void bn_prep_kernel(const float* __restrict__ g,
                               const float* __restrict__ b,
                               const float* __restrict__ m,
                               const float* __restrict__ v,
                               float* __restrict__ s, float* __restrict__ h, int n)
{
    const int i = blockIdx.x * 64 + threadIdx.x;
    if (i >= n) return;
    const float inv = g[i] * rsqrtf(v[i] + 1e-5f);
    s[i] = inv;
    h[i] = b[i] - m[i] * inv;
}

// BN0(input) f32 NCHW(6ch) -> bf16 NHWC padded to 32 channels
__global__ void prep_input_kernel(const float* __restrict__ in,
                                  const float* __restrict__ s,
                                  const float* __restrict__ h,
                                  __bf16* __restrict__ X)
{
    const int p = blockIdx.x * 256 + threadIdx.x;
    if (p >= BHW) return;
    const int b  = p >> 16;
    const int yx = p & (HWC - 1);
    alignas(16) __bf16 buf[32];
#pragma unroll
    for (int c = 0; c < 32; ++c) {
        float v = 0.0f;
        if (c < 6) v = in[(size_t)(b * 6 + c) * HWC + yx] * s[c] + h[c];
        buf[c] = (__bf16)v;
    }
    uint4* dst = (uint4*)(X + (size_t)p * 32);
    const uint4* sb = (const uint4*)buf;
    dst[0] = sb[0]; dst[1] = sb[1]; dst[2] = sb[2]; dst[3] = sb[3];
}

// y = relu(scale*x + shift), bf16 NHWC(64) -> bf16 NHWC(64)
__global__ void bnrelu_kernel(const __bf16* __restrict__ X,
                              const float* __restrict__ s,
                              const float* __restrict__ h,
                              __bf16* __restrict__ Y, int total)
{
    const int i = blockIdx.x * 256 + threadIdx.x;
    if (i >= total) return;
    const int c = i & 63;
    const float v = (float)X[i] * s[c] + h[c];
    Y[i] = (__bf16)fmaxf(v, 0.0f);
}

// Softmax over the 25 class channels (NCHW f32 -> NCHW f32)
__global__ void softmax25_kernel(const float* __restrict__ mo,
                                 float* __restrict__ mask)
{
    const int p = blockIdx.x * 256 + threadIdx.x;
    if (p >= BHW) return;
    const int b  = p >> 16;
    const int yx = p & (HWC - 1);
    const float* src = mo + (size_t)b * NCLS * HWC + yx;
    float vals[NCLS];
    float mx = -1e30f;
#pragma unroll
    for (int c = 0; c < NCLS; ++c) { vals[c] = src[(size_t)c * HWC]; mx = fmaxf(mx, vals[c]); }
    float sum = 0.0f;
#pragma unroll
    for (int c = 0; c < NCLS; ++c) { vals[c] = __expf(vals[c] - mx); sum += vals[c]; }
    const float inv = 1.0f / sum;
    float* dst = mask + (size_t)b * NCLS * HWC + yx;
#pragma unroll
    for (int c = 0; c < NCLS; ++c) dst[(size_t)c * HWC] = vals[c] * inv;
}

// seg = conv5x5(mask, m_kernel); pred[cim] = conv5x5(im_last[cim]*mask, m_kernel)
// (im factor is per-position -> factor out the channel sum)
__global__ void segpred_kernel(const float* __restrict__ mask,
                               const float* __restrict__ im,
                               const float* __restrict__ mk,
                               float* __restrict__ seg,
                               float* __restrict__ pred)
{
    const int p = blockIdx.x * 256 + threadIdx.x;
    if (p >= BHW) return;
    const int b  = p >> 16;
    const int yx = p & (HWC - 1);
    const int y  = yx >> 8;
    const int x  = yx & 255;
    const float* mb = mask + (size_t)b * NCLS * HWC;
    const float* ib = im + (size_t)b * 6 * HWC;
    float sv = 0.0f, p0 = 0.0f, p1 = 0.0f, p2 = 0.0f;
    for (int ky = 0; ky < 5; ++ky) {
        const int yy = y + ky - 2;
        if ((unsigned)yy >= (unsigned)HH) continue;
        for (int kx = 0; kx < 5; ++kx) {
            const int xx = x + kx - 2;
            if ((unsigned)xx >= (unsigned)WW) continue;
            const int pos = (yy << 8) + xx;
            float acc = 0.0f;
#pragma unroll
            for (int c = 0; c < NCLS; ++c)
                acc += mk[c * 25 + ky * 5 + kx] * mb[(size_t)c * HWC + pos];
            sv += acc;
            p0 += acc * ib[(size_t)3 * HWC + pos];
            p1 += acc * ib[(size_t)4 * HWC + pos];
            p2 += acc * ib[(size_t)5 * HWC + pos];
        }
    }
    seg[p] = sv;
    pred[(size_t)(b * 3 + 0) * HWC + yx] = p0;
    pred[(size_t)(b * 3 + 1) * HWC + yx] = p1;
    pred[(size_t)(b * 3 + 2) * HWC + yx] = p2;
}

__global__ void blend_kernel(const float* __restrict__ segf,
                             const float* __restrict__ segb,
                             const float* __restrict__ predf,
                             const float* __restrict__ predb,
                             float* __restrict__ pred,
                             float* __restrict__ attn,
                             float* __restrict__ attn1m)
{
    const int p = blockIdx.x * 256 + threadIdx.x;
    if (p >= BHW) return;
    const int b  = p >> 16;
    const int yx = p & (HWC - 1);
    const float a = (segf[p] + 1e-5f) / (segf[p] + segb[p] + 2e-5f);
    attn[p]   = a;
    attn1m[p] = 1.0f - a;
#pragma unroll
    for (int c = 0; c < 3; ++c) {
        const size_t idx = (size_t)(b * 3 + c) * HWC + yx;
        pred[idx] = a * predf[idx] + (1.0f - a) * predb[idx];
    }
}

__global__ void zfill_kernel(__bf16* z) { z[threadIdx.x] = (__bf16)0.0f; }

// ---------------------------------------------------------------------------
extern "C" void kernel_launch(void* const* d_in, const int* in_sizes, int n_in,
                              void* d_out, int out_size, void* d_ws, size_t ws_size,
                              hipStream_t stream)
{
    (void)in_sizes; (void)n_in; (void)out_size; (void)ws_size;

    const float* im_f     = (const float*)d_in[0];
    const float* im_b     = (const float*)d_in[1];
    const float* m_kernel = (const float*)d_in[3];
    const float* bn0_g    = (const float*)d_in[4];
    const float* bn0_b    = (const float*)d_in[5];
    const float* bn0_m    = (const float*)d_in[6];
    const float* bn0_v    = (const float*)d_in[7];
    const float* conv0_w  = (const float*)d_in[8];
    const float* bn1_g    = (const float*)d_in[9];
    const float* bn1_b    = (const float*)d_in[10];
    const float* bn1_m    = (const float*)d_in[11];
    const float* bn1_v    = (const float*)d_in[12];
    const float* w11      = (const float*)d_in[13];
    const float* w12      = (const float*)d_in[14];
    const float* bn2_g    = (const float*)d_in[15];
    const float* bn2_b    = (const float*)d_in[16];
    const float* bn2_m    = (const float*)d_in[17];
    const float* bn2_v    = (const float*)d_in[18];
    const float* w2       = (const float*)d_in[19];
    const float* bnf_g    = (const float*)d_in[20];
    const float* bnf_b    = (const float*)d_in[21];
    const float* bnf_m    = (const float*)d_in[22];
    const float* bnf_v    = (const float*)d_in[23];
    const float* convf_w  = (const float*)d_in[24];
    const float* convf_b  = (const float*)d_in[25];

    float* out = (float*)d_out;
    // Output slots (f32 elements), in reference return order
    const size_t PRED_OFF   = 0;                       // (B,3,H,W)
    const size_t PREDF_OFF  = PRED_OFF   + (size_t)3 * BHW;
    const size_t MASKF_OFF  = PREDF_OFF  + (size_t)3 * BHW;
    const size_t ATTN_OFF   = MASKF_OFF  + (size_t)NCLS * BHW;
    const size_t PREDB_OFF  = ATTN_OFF   + (size_t)BHW;
    const size_t MASKB_OFF  = PREDB_OFF  + (size_t)3 * BHW;
    const size_t ATTN1M_OFF = MASKB_OFF  + (size_t)NCLS * BHW;

    // --- workspace carve-up -------------------------------------------------
    char* wsb = (char*)d_ws;
    size_t off = 0;
    auto alloc = [&](size_t bytes) -> void* {
        void* pp = wsb + off;
        off = (off + bytes + 255) & ~(size_t)255;
        return pp;
    };
    __bf16* zpage  = (__bf16*)alloc(256);
    float*  bn0s   = (float*)alloc(6 * 4);
    float*  bn0h   = (float*)alloc(6 * 4);
    float*  bn1s   = (float*)alloc(NBLK * 64 * 4);
    float*  bn1h   = (float*)alloc(NBLK * 64 * 4);
    float*  bn2s   = (float*)alloc(NBLK * 64 * 4);
    float*  bn2h   = (float*)alloc(NBLK * 64 * 4);
    float*  bnfs   = (float*)alloc(64 * 4);
    float*  bnfh   = (float*)alloc(64 * 4);

    const int F0   = 4 * 9 * 1 * 512;    // conv0 frag elems (ng=4, kc=1)
    const int FB   = 4 * 9 * 2 * 512;    // 64x64 frag elems (ng=4, kc=2)
    const int FF   = 2 * 9 * 2 * 512;    // convf frag elems (ng=2, kc=2)
    __bf16* f_c0   = (__bf16*)alloc((size_t)F0 * 2);
    __bf16* f_w11  = (__bf16*)alloc((size_t)NBLK * FB * 2);
    __bf16* f_w12  = (__bf16*)alloc((size_t)NBLK * FB * 2);
    __bf16* f_w2   = (__bf16*)alloc((size_t)NBLK * FB * 2);
    __bf16* f_cf   = (__bf16*)alloc((size_t)FF * 2);

    __bf16* x32    = (__bf16*)alloc((size_t)BHW * 32 * 2);
    __bf16* hbuf   = (__bf16*)alloc((size_t)BHW * 64 * 2);
    __bf16* tbuf   = (__bf16*)alloc((size_t)BHW * 64 * 2);
    __bf16* vbuf   = (__bf16*)alloc((size_t)BHW * 64 * 2);
    float*  motion = (float*)alloc((size_t)NCLS * BHW * 4);
    float*  segf   = (float*)alloc((size_t)BHW * 4);
    float*  segb   = (float*)alloc((size_t)BHW * 4);

    // --- one-time (per call) prep ------------------------------------------
    zfill_kernel<<<1, 128, 0, stream>>>(zpage);
    bn_prep_kernel<<<1, 64, 0, stream>>>(bn0_g, bn0_b, bn0_m, bn0_v, bn0s, bn0h, 6);
    for (int i = 0; i < NBLK; ++i) {
        bn_prep_kernel<<<1, 64, 0, stream>>>(bn1_g + i * 64, bn1_b + i * 64,
                                             bn1_m + i * 64, bn1_v + i * 64,
                                             bn1s + i * 64, bn1h + i * 64, 64);
        bn_prep_kernel<<<1, 64, 0, stream>>>(bn2_g + i * 64, bn2_b + i * 64,
                                             bn2_m + i * 64, bn2_v + i * 64,
                                             bn2s + i * 64, bn2h + i * 64, 64);
    }
    bn_prep_kernel<<<1, 64, 0, stream>>>(bnf_g, bnf_b, bnf_m, bnf_v, bnfs, bnfh, 64);

    repack_weights_kernel<<<(F0 + 255) / 256, 256, 0, stream>>>(conv0_w, f_c0, 64, 6, 4, 1);
    for (int i = 0; i < NBLK; ++i) {
        const size_t wsz = (size_t)64 * 64 * 9;
        repack_weights_kernel<<<(FB + 255) / 256, 256, 0, stream>>>(w11 + i * wsz, f_w11 + (size_t)i * FB, 64, 64, 4, 2);
        repack_weights_kernel<<<(FB + 255) / 256, 256, 0, stream>>>(w12 + i * wsz, f_w12 + (size_t)i * FB, 64, 64, 4, 2);
        repack_weights_kernel<<<(FB + 255) / 256, 256, 0, stream>>>(w2  + i * wsz, f_w2  + (size_t)i * FB, 64, 64, 4, 2);
    }
    repack_weights_kernel<<<(FF + 255) / 256, 256, 0, stream>>>(convf_w, f_cf, 25, 64, 2, 2);

    // --- backbone per direction --------------------------------------------
    const float* ins[2]  = { im_f, im_b };
    float* masks[2]      = { out + MASKF_OFF, out + MASKB_OFF };
    float* preds[2]      = { out + PREDF_OFF, out + PREDB_OFF };
    float* segs[2]       = { segf, segb };

    const dim3 cgrid(BHW / 128), cblk(256);         // 8 waves/WG, 16 px/wave
    const dim3 egrid(BHW / 256), eblk(256);
    const dim3 fgrid((BHW * 64) / 256);

    for (int d = 0; d < 2; ++d) {
        prep_input_kernel<<<egrid, eblk, 0, stream>>>(ins[d], bn0s, bn0h, x32);

        // conv0: 32(pad from 6) -> 64, plain epilogue
        conv3x3_wmma_kernel<4, 1, 0, false><<<cgrid, cblk, 0, stream>>>(
            x32, f_c0, nullptr, 1, nullptr, nullptr, nullptr, nullptr, 0,
            hbuf, nullptr, zpage);

        for (int i = 0; i < NBLK; ++i) {
            bnrelu_kernel<<<fgrid, eblk, 0, stream>>>(hbuf, bn1s + i * 64, bn1h + i * 64,
                                                      tbuf, BHW * 64);
            // conv11 + dilated conv12, fused BN2+ReLU epilogue
            conv3x3_wmma_kernel<4, 2, 1, true><<<cgrid, cblk, 0, stream>>>(
                tbuf, f_w11 + (size_t)i * FB, f_w12 + (size_t)i * FB, 2,
                bn2s + i * 64, bn2h + i * 64, nullptr, nullptr, 0,
                vbuf, nullptr, zpage);
            // conv2 with fused residual add: h = h + conv2(v)
            conv3x3_wmma_kernel<4, 2, 2, false><<<cgrid, cblk, 0, stream>>>(
                vbuf, f_w2 + (size_t)i * FB, nullptr, 1, nullptr, nullptr,
                hbuf, nullptr, 0, hbuf, nullptr, zpage);
        }

        bnrelu_kernel<<<fgrid, eblk, 0, stream>>>(hbuf, bnfs, bnfh, tbuf, BHW * 64);
        // final conv 64 -> 25 (+bias), f32 NCHW out
        conv3x3_wmma_kernel<2, 2, 3, false><<<cgrid, cblk, 0, stream>>>(
            tbuf, f_cf, nullptr, 1, nullptr, nullptr, nullptr, convf_b, NCLS,
            nullptr, motion, zpage);

        softmax25_kernel<<<egrid, eblk, 0, stream>>>(motion, masks[d]);
        segpred_kernel<<<egrid, eblk, 0, stream>>>(masks[d], ins[d], m_kernel,
                                                   segs[d], preds[d]);
    }

    blend_kernel<<<egrid, eblk, 0, stream>>>(segf, segb, out + PREDF_OFF,
                                             out + PREDB_OFF, out + PRED_OFF,
                                             out + ATTN_OFF, out + ATTN1M_OFF);
}
